// SelfAttention_61847529062712
// MI455X (gfx1250) — matrix operand order repaired and checked by reading, not compiled, not run on hardware
//
#include <hip/hip_runtime.h>

// ---------------------------------------------------------------------------
// MI455X (gfx1250) self-attention, f16 WMMA pipeline with flash-attention core
// and async global->LDS staging (ASYNCcnt path).
//   B=8, C=512, H=W=64, N=4096, D=64
// Workspace layout (f16): Qh | Kh | Vh | Oh, each B*N*D = 2M elems (4 MiB).
// ---------------------------------------------------------------------------

typedef __attribute__((ext_vector_type(16))) _Float16 v16h;
typedef __attribute__((ext_vector_type(8)))  _Float16 v8h;
typedef __attribute__((ext_vector_type(8)))  float    v8f;

#define BATCH 8
#define CCH   512
#define NSEQ  4096
#define DDIM  64

// ---- WMMA operand loaders (wave32; contiguous k order, same for A and B) -----
// A matrix 16x32 f16: lane holds row (lane&15); element i = K kb+i
// with kb = 16*(lane>=16). Two 16-byte loads.
__device__ __forceinline__ v16h ld_a_tile(const _Float16* base, int ld, int kofs) {
  const int lane = threadIdx.x & 31;
  const _Float16* rp = base + (lane & 15) * ld + kofs + ((lane & 16) ? 16 : 0);
  v8h lo = *(const v8h*)(rp);
  v8h hi = *(const v8h*)(rp + 8);
  v16h r;
#pragma unroll
  for (int i = 0; i < 8; ++i) { r[i] = lo[i]; r[i + 8] = hi[i]; }
  return r;
}

// B matrix 32x16 f16 loaded from a B-transposed image (rows = output column,
// contraction contiguous): lane holds col (lane&15); element i = K kb+i with
// kb = 16*(lane>=16). 32 contiguous bytes per lane.
__device__ __forceinline__ v16h ld_bt_tile(const _Float16* base, int ld, int kofs) {
  const int lane = threadIdx.x & 31;
  const _Float16* rp = base + (lane & 15) * ld + kofs + ((lane & 16) ? 16 : 0);
  v8h lo = *(const v8h*)(rp);
  v8h hi = *(const v8h*)(rp + 8);
  v16h r;
#pragma unroll
  for (int i = 0; i < 8; ++i) { r[i] = lo[i]; r[i + 8] = hi[i]; }
  return r;
}

__device__ __forceinline__ v8f wmma_f16(v16h a, v16h b, v8f c) {
  // (neg_a, A, neg_b, B, c_mod, C, reuse_a, reuse_b)
  return __builtin_amdgcn_wmma_f32_16x16x32_f16(false, a, false, b, (short)0, c,
                                                false, false);
}

// Async contiguous 16B global->LDS copy (GV mode): LDS byte address taken from %0,
// tracked by ASYNCcnt. CDNA5 ISA / CDNA5 ISA.
__device__ __forceinline__ void async_g2l_b128(unsigned lds_byte_off,
                                               const void* gaddr) {
  unsigned long long ga = (unsigned long long)gaddr;
  asm volatile("global_load_async_to_lds_b128 %0, %1, off"
               :: "v"(lds_byte_off), "v"(ga) : "memory");
}
__device__ __forceinline__ void wait_async0() {
  asm volatile("s_wait_asynccnt 0x0" ::: "memory");
}

// ---------------------------------------------------------------------------
// Kernel 1: Q/K/V projections. blockIdx.z selects which projection.
//   Out[b,n,d] = sum_c W[d,c] * x[b,c,n] + bias[d]   (f16 output, (B,N,D))
// Block = 128 threads (4 waves); tile = 64 n-rows x 64 d-cols; K chunks of 32.
// Both stagings convert f32->f16, so they must pass through VALU (no async).
// ---------------------------------------------------------------------------
__global__ __launch_bounds__(128) void qkv_proj_kernel(
    const float* __restrict__ x,
    const float* __restrict__ Wq, const float* __restrict__ bq,
    const float* __restrict__ Wk, const float* __restrict__ bk,
    const float* __restrict__ Wv, const float* __restrict__ bv,
    _Float16* __restrict__ Qh, _Float16* __restrict__ Kh,
    _Float16* __restrict__ Vh)
{
  const int b = blockIdx.y;
  const int ntile = blockIdx.x * 64;
  const int which = blockIdx.z;
  const float* W  = (which == 0) ? Wq : (which == 1) ? Wk : Wv;
  const float* bs = (which == 0) ? bq : (which == 1) ? bk : bv;
  _Float16* Out   = (which == 0) ? Qh : (which == 1) ? Kh : Vh;

  __shared__ __align__(16) _Float16 Al[64 * 32];  // A tile: [n][c]  (x^T, f16)
  __shared__ __align__(16) _Float16 Bl[64 * 32];  // B^T tile: [d][c] (W, f16)

  const int w = threadIdx.x >> 5;
  const int lane = threadIdx.x & 31;

  v8f zero = {};
  v8f acc[4];
#pragma unroll
  for (int t = 0; t < 4; ++t) acc[t] = zero;

  for (int c0 = 0; c0 < CCH; c0 += 32) {
    // Stage x transposed: Al[n][c] = x[b, c0+c, ntile+n] (coalesced in n)
    for (int t = threadIdx.x; t < 2048; t += 128) {
      int c = t >> 6, n = t & 63;
      float xv = x[((size_t)b * CCH + (c0 + c)) * NSEQ + ntile + n];
      Al[n * 32 + c] = (_Float16)xv;
    }
    // Stage weights: Bl[d][c] = W[d, c0+c] (natural row-major = B^T)
    for (int t = threadIdx.x; t < 2048; t += 128) {
      int d = t >> 5, c = t & 31;
      Bl[d * 32 + c] = (_Float16)W[d * CCH + c0 + c];
    }
    __syncthreads();

    v16h a = ld_a_tile(Al + w * 16 * 32, 32, 0);
#pragma unroll
    for (int t = 0; t < 4; ++t) {
      v16h bm = ld_bt_tile(Bl + t * 16 * 32, 32, 0);
      acc[t] = wmma_f16(a, bm, acc[t]);
    }
    __syncthreads();
  }

#pragma unroll
  for (int t = 0; t < 4; ++t)
#pragma unroll
    for (int r = 0; r < 8; ++r) {
      int row = r + ((lane & 16) ? 8 : 0);
      int n = ntile + w * 16 + row;
      int d = t * 16 + (lane & 15);
      float v = acc[t][r] + bs[d];
      Out[((size_t)b * NSEQ + n) * DDIM + d] = (_Float16)v;
    }
}

// ---------------------------------------------------------------------------
// Kernel 2: flash attention. 4 waves/block; 64 query rows per block (16/wave);
// key tiles of 64 staged in LDS; online softmax in f32; S and P never reach
// HBM (would be 537 MB/pass otherwise).
// Dynamic LDS layout (base offset 0, no static allocs):
//   [    0, 8192)  Kt[key][d]          (async-copied, contiguous)
//   [ 8192,16384)  Vt[d][key]          (VGPR-mediated transpose store)
//   [16384,24576)  Pl[wave][m][key]    (P repack bounce)
// ---------------------------------------------------------------------------
__global__ __launch_bounds__(128) void flash_attn_kernel(
    const _Float16* __restrict__ Qh, const _Float16* __restrict__ Kh,
    const _Float16* __restrict__ Vh, _Float16* __restrict__ Oh)
{
  extern __shared__ __align__(16) _Float16 smem[];
  _Float16* Kt = smem;               // 4096 elems
  _Float16* Vt = smem + 4096;        // 4096 elems
  _Float16* Pl = smem + 8192;        // 4 * 1024 elems

  const int b = blockIdx.y;
  const int qtile = blockIdx.x * 64;
  const int w = threadIdx.x >> 5;
  const int lane = threadIdx.x & 31;

  // Q rows for this wave, loaded once directly in A layout (rows stride 128B)
  const _Float16* Qb = Qh + ((size_t)b * NSEQ + qtile + w * 16) * DDIM;
  v16h qa0 = ld_a_tile(Qb, DDIM, 0);
  v16h qa1 = ld_a_tile(Qb, DDIM, 32);

  float m[8], l[8];
  v8f zero = {};
  v8f o[4];
#pragma unroll
  for (int r = 0; r < 8; ++r) { m[r] = -3.0e38f; l[r] = 0.f; }
#pragma unroll
  for (int t = 0; t < 4; ++t) o[t] = zero;

  for (int j = 0; j < NSEQ; j += 64) {
    // K tile: straight 8 KB copy -> async global->LDS (ASYNCcnt), 4 ops/lane
    const _Float16* ksrc = Kh + ((size_t)b * NSEQ + j) * DDIM;
    for (int t = threadIdx.x; t < 512; t += 128)
      async_g2l_b128((unsigned)(t * 16), ksrc + (size_t)t * 8);

    // V tile needs a 2-byte-granularity transpose -> VGPR-mediated staging
    for (int t = threadIdx.x; t < 4096; t += 128) {
      int key = t >> 6, d = t & 63;
      Vt[d * 64 + key] = Vh[((size_t)b * NSEQ + j + key) * DDIM + d];
    }
    if (j + 64 < NSEQ)  // hint next K tile toward the caches
      __builtin_prefetch(Kh + ((size_t)b * NSEQ + j + 64) * DDIM +
                             (size_t)threadIdx.x * 32, 0, 1);
    wait_async0();
    __syncthreads();

    // S = Q K^T : 16x64 per wave, contraction d=64 in two 32-chunks
    v8f s[4];
#pragma unroll
    for (int t = 0; t < 4; ++t) {
      v16h b0 = ld_bt_tile(Kt + t * 16 * 64, 64, 0);
      v16h b1 = ld_bt_tile(Kt + t * 16 * 64, 64, 32);
      s[t] = wmma_f16(qa0, b0, zero);
      s[t] = wmma_f16(qa1, b1, s[t]);
    }

    // ---- online softmax (rows live in 16-lane halves of the C layout) ----
    float mn[8], rs[8];
#pragma unroll
    for (int r = 0; r < 8; ++r)
      mn[r] = fmaxf(fmaxf(s[0][r], s[1][r]), fmaxf(s[2][r], s[3][r]));
#pragma unroll
    for (int off = 8; off >= 1; off >>= 1)
#pragma unroll
      for (int r = 0; r < 8; ++r)
        mn[r] = fmaxf(mn[r], __shfl_xor(mn[r], off, 32));
#pragma unroll
    for (int r = 0; r < 8; ++r) {
      float mnew = fmaxf(m[r], mn[r]);
      float sc = __expf(m[r] - mnew);
      m[r] = mnew;
      l[r] *= sc;
      float sum = 0.f;
#pragma unroll
      for (int t = 0; t < 4; ++t) {
        float p = __expf(s[t][r] - mnew);
        s[t][r] = p;
        sum += p;
        o[t][r] *= sc;
      }
      rs[r] = sum;
    }
#pragma unroll
    for (int off = 8; off >= 1; off >>= 1)
#pragma unroll
      for (int r = 0; r < 8; ++r)
        rs[r] += __shfl_xor(rs[r], off, 32);
#pragma unroll
    for (int r = 0; r < 8; ++r) l[r] += rs[r];

    // Repack P (C layout -> A layout) through a wave-private LDS bounce.
    _Float16* P = Pl + w * 1024;
#pragma unroll
    for (int t = 0; t < 4; ++t)
#pragma unroll
      for (int r = 0; r < 8; ++r) {
        int row = r + ((lane & 16) ? 8 : 0);
        P[row * 64 + t * 16 + (lane & 15)] = (_Float16)s[t][r];
      }
    v16h pa0 = ld_a_tile(P, 64, 0);   // keys 0..31
    v16h pa1 = ld_a_tile(P, 64, 32);  // keys 32..63

    // O += P * V : contraction over 64 keys in two 32-chunks
#pragma unroll
    for (int t = 0; t < 4; ++t) {
      v16h vb0 = ld_bt_tile(Vt + t * 16 * 64, 64, 0);
      v16h vb1 = ld_bt_tile(Vt + t * 16 * 64, 64, 32);
      o[t] = wmma_f16(pa0, vb0, o[t]);
      o[t] = wmma_f16(pa1, vb1, o[t]);
    }
    __syncthreads();
  }

  // Normalize and store O (B,N,D) f16
#pragma unroll
  for (int t = 0; t < 4; ++t)
#pragma unroll
    for (int r = 0; r < 8; ++r) {
      int row = r + ((lane & 16) ? 8 : 0);
      int n = qtile + w * 16 + row;
      float v = o[t][r] / l[r];
      Oh[((size_t)b * NSEQ + n) * DDIM + t * 16 + (lane & 15)] = (_Float16)v;
    }
}

// ---------------------------------------------------------------------------
// Kernel 3: output projection + residual.
//   out[b,c,n] = gamma * (sum_d Wo[c,d] * O[b,n,d] + bo[c]) + x[b,c,n]
// Tile 64c x 64n per block; contraction d=64 fully staged (no K loop).
// Dynamic LDS: [0,8192) Wl[c][d] (converted f32->f16), [8192,16384) Ol[n][d]
// (async-copied, contiguous f16).
// ---------------------------------------------------------------------------
__global__ __launch_bounds__(128) void out_proj_kernel(
    const _Float16* __restrict__ Oh, const float* __restrict__ Wo,
    const float* __restrict__ bo, const float* __restrict__ gamma,
    const float* __restrict__ x, float* __restrict__ out)
{
  extern __shared__ __align__(16) _Float16 smem[];
  _Float16* Wl = smem;          // 4096 elems: A tile [c][d]
  _Float16* Ol = smem + 4096;   // 4096 elems: B^T tile [n][d]

  const int b = blockIdx.z;
  const int ctile = blockIdx.y * 64;
  const int ntile = blockIdx.x * 64;
  const int w = threadIdx.x >> 5;
  const int lane = threadIdx.x & 31;

  // O tile: contiguous f16 -> async global->LDS copy
  const _Float16* osrc = Oh + ((size_t)b * NSEQ + ntile) * DDIM;
  for (int t = threadIdx.x; t < 512; t += 128)
    async_g2l_b128((unsigned)(8192 + t * 16), osrc + (size_t)t * 8);

  // Wo tile: f32 -> f16 conversion through VALU
  for (int t = threadIdx.x; t < 4096; t += 128) {
    int c = t >> 6, d = t & 63;
    Wl[t] = (_Float16)Wo[(ctile + c) * DDIM + d];
  }
  wait_async0();
  __syncthreads();

  v8f zero = {};
  v8f acc[4];
#pragma unroll
  for (int t = 0; t < 4; ++t) acc[t] = zero;

  v16h a0 = ld_a_tile(Wl + w * 16 * 64, 64, 0);
  v16h a1 = ld_a_tile(Wl + w * 16 * 64, 64, 32);
#pragma unroll
  for (int t = 0; t < 4; ++t) {
    v16h b0 = ld_bt_tile(Ol + t * 16 * 64, 64, 0);
    v16h b1 = ld_bt_tile(Ol + t * 16 * 64, 64, 32);
    acc[t] = wmma_f16(a0, b0, acc[t]);
    acc[t] = wmma_f16(a1, b1, acc[t]);
  }

  const float g = gamma[0];
#pragma unroll
  for (int t = 0; t < 4; ++t)
#pragma unroll
    for (int r = 0; r < 8; ++r) {
      int row = r + ((lane & 16) ? 8 : 0);
      int c = ctile + w * 16 + row;
      int n = ntile + t * 16 + (lane & 15);
      size_t idx = ((size_t)b * CCH + c) * NSEQ + n;
      out[idx] = g * (acc[t][r] + bo[c]) + x[idx];
    }
}

// ---------------------------------------------------------------------------
extern "C" void kernel_launch(void* const* d_in, const int* in_sizes, int n_in,
                              void* d_out, int out_size, void* d_ws, size_t ws_size,
                              hipStream_t stream) {
  const float* x     = (const float*)d_in[0];
  const float* Wq    = (const float*)d_in[1];
  const float* bq    = (const float*)d_in[2];
  const float* Wk    = (const float*)d_in[3];
  const float* bk    = (const float*)d_in[4];
  const float* Wv    = (const float*)d_in[5];
  const float* bv    = (const float*)d_in[6];
  const float* Wo    = (const float*)d_in[7];
  const float* bo    = (const float*)d_in[8];
  const float* gamma = (const float*)d_in[9];
  float* out = (float*)d_out;

  const size_t elems = (size_t)BATCH * NSEQ * DDIM;  // 2M f16 = 4 MiB each
  _Float16* Qh = (_Float16*)d_ws;
  _Float16* Kh = Qh + elems;
  _Float16* Vh = Kh + elems;
  _Float16* Oh = Vh + elems;

  // 1) Q/K/V projections: grid z selects projection; 64-row n-tiles per block
  qkv_proj_kernel<<<dim3(NSEQ / 64, BATCH, 3), 128, 0, stream>>>(
      x, Wq, bq, Wk, bk, Wv, bv, Qh, Kh, Vh);

  // 2) Flash attention: one block per 64 query rows per batch (24 KB dyn LDS)
  flash_attn_kernel<<<dim3(NSEQ / 64, BATCH), 128, 24576, stream>>>(
      Qh, Kh, Vh, Oh);

  // 3) Output projection + gamma + residual (16 KB dyn LDS)
  out_proj_kernel<<<dim3(NSEQ / 64, CCH / 64, BATCH), 128, 16384, stream>>>(
      Oh, Wo, bo, gamma, x, out);
}
